// DQN_41171556500061
// MI455X (gfx1250) — matrix operand order repaired and checked by reading, not compiled, not run on hardware
//
#include <hip/hip_runtime.h>
#include <hip/hip_bf16.h>
#include <math.h>

typedef _Float16 h16;
typedef __attribute__((ext_vector_type(16))) _Float16 v16h;
typedef __attribute__((ext_vector_type(8)))  _Float16 v8h;
typedef __attribute__((ext_vector_type(8)))  float    v8f;

#define N_OBS 100
#define N_ACT 20
#define HDIM  128
#define KPAD  128
#define WAVES 8
#define ROWS_PER_WAVE 32            // two 16-row WMMA tiles per wave -> 2 indep. chains
#define ROWS_PER_BLOCK (WAVES * ROWS_PER_WAVE)   // 256

// ---- LDS layout (bytes) ----
#define OFF_W1T   0                      // h16 [128][128] = 32768
#define OFF_W2T   32768                  // h16 [128][128] = 32768
#define OFF_W3T   65536                  // h16 [ 32][128] =  8192
#define OFF_B1    73728                  // f32 [128]      =   512
#define OFF_B2    74240                  // f32 [128]      =   512
#define OFF_B3    74752                  // f32 [ 32]      =   128
#define OFF_X     74880                  // h16 [8][32][128] = 65536
#define OFF_H     140416                 // h16 [8][32][128] = 65536
#define OFF_LG    OFF_H                  // f32 [8][32][32]  = 32768 (aliases sH; dead by layer3)
#define OFF_GS    205952                 // f32 [8][32][20]  = 20480
#define LDS_TOTAL 226432

// A fragment, 16-bit 16x32 (MxK), wave32:
//   lanes 0-15: M=lane,    halves e=0..15 -> K=k0+e
//   lanes16-31: M=lane-16, halves e=0..15 -> K=k0+16+e
__device__ __forceinline__ v16h load_a_frag(const h16* tile, int m, int g, int k0) {
    const v8h* p0 = (const v8h*)(tile + m * KPAD + k0 + 16 * g);
    v8h lo = p0[0];
    v8h hi = p0[1];  // +8 halves
    v16h a;
#pragma unroll
    for (int e = 0; e < 8; ++e) { a[e] = lo[e]; a[e + 8] = hi[e]; }
    return a;
}

// B fragment, 16-bit 32x16 (KxN), wave32, weights stored transposed [N][KPAD]:
//   lanes 0-15: N=lane,    halves e=0..15 -> K=k0+e
//   lanes16-31: N=lane-16, halves e=0..15 -> K=k0+16+e
__device__ __forceinline__ v16h load_b_frag(const h16* wt, int n, int g, int k0) {
    const v8h* p = (const v8h*)(wt + n * KPAD + k0 + 16 * g);
    v8h lo = p[0];
    v8h hi = p[1];
    v16h b;
#pragma unroll
    for (int e = 0; e < 8; ++e) { b[e] = lo[e]; b[e + 8] = hi[e]; }
    return b;
}

// Hidden layer for one wave's 32-row tile: dual independent WMMA chains sharing B
// fragments, software-pipelined B loads, bias + leaky_relu(0.4), f16 out.
__device__ __forceinline__ void gemm_dual_hidden(const h16* inT, const h16* WT,
                                                 const float* bias, h16* outT, int lane) {
    const int g  = lane >> 4;
    const int ln = lane & 15;

    v16h a0[4], a1[4];
#pragma unroll
    for (int k = 0; k < 4; ++k) {
        a0[k] = load_a_frag(inT, ln, g, 32 * k);
        a1[k] = load_a_frag(inT + 16 * KPAD, ln, g, 32 * k);
    }

    for (int j = 0; j < HDIM / 16; ++j) {
        v8f  c0 = {}, c1 = {};
        v16h bc = load_b_frag(WT, j * 16 + ln, g, 0);
#pragma unroll
        for (int k = 0; k < 4; ++k) {
            v16h bn = bc;
            if (k < 3) bn = load_b_frag(WT, j * 16 + ln, g, 32 * (k + 1));  // prefetch
            c0 = __builtin_amdgcn_wmma_f32_16x16x32_f16(false, a0[k], false, bc, (short)0,
                                                        c0, false, false);
            c1 = __builtin_amdgcn_wmma_f32_16x16x32_f16(false, a1[k], false, bc, (short)0,
                                                        c1, false, false);
            bc = bn;
        }
        const int   N  = j * 16 + ln;
        const float bv = bias[N];
#pragma unroll
        for (int r = 0; r < 8; ++r) {
            const int M = r + 8 * g;            // C/D layout: M = r + 8*(lane>>4)
            float v0 = c0[r] + bv;
            float v1 = c1[r] + bv;
            v0 = (v0 < 0.0f) ? 0.4f * v0 : v0;  // leaky_relu slope 0.4
            v1 = (v1 < 0.0f) ? 0.4f * v1 : v1;
            outT[M * KPAD + N]        = (h16)v0;
            outT[(16 + M) * KPAD + N] = (h16)v1;
        }
    }
}

__global__ __launch_bounds__(256) void dqn_fused(
    const float* __restrict__ x,
    const float* __restrict__ W1, const float* __restrict__ b1,
    const float* __restrict__ W2, const float* __restrict__ b2,
    const float* __restrict__ W3, const float* __restrict__ b3,
    float* __restrict__ out, int B) {
    extern __shared__ __align__(16) char smem[];
    h16*   sW1T = (h16*)(smem + OFF_W1T);
    h16*   sW2T = (h16*)(smem + OFF_W2T);
    h16*   sW3T = (h16*)(smem + OFF_W3T);
    float* sB1  = (float*)(smem + OFF_B1);
    float* sB2  = (float*)(smem + OFF_B2);
    float* sB3  = (float*)(smem + OFF_B3);
    h16*   sX   = (h16*)(smem + OFF_X);
    h16*   sH   = (h16*)(smem + OFF_H);
    float* sLg  = (float*)(smem + OFF_LG);   // aliases sH (dead after layer-2 reads)
    float* sGs  = (float*)(smem + OFF_GS);

    const int tid = threadIdx.x;

    // ---- stage weights once per block: f32 -> f16, transposed [N][KPAD], zero-padded K ----
    for (int i = tid; i < HDIM * KPAD; i += 256) {
        const int n = i >> 7, k = i & 127;
        sW1T[i] = (h16)((k < N_OBS) ? W1[k * HDIM + n] : 0.0f);
        sW2T[i] = (h16)(W2[k * HDIM + n]);
    }
    for (int i = tid; i < 32 * KPAD; i += 256) {
        const int n = i >> 7, k = i & 127;
        sW3T[i] = (h16)((n < N_ACT) ? W3[k * N_ACT + n] : 0.0f);
    }
    for (int i = tid; i < HDIM; i += 256) { sB1[i] = b1[i]; sB2[i] = b2[i]; }
    if (tid < 32) sB3[tid] = (tid < N_ACT) ? b3[tid] : 0.0f;

    const int  wave    = tid >> 5;
    const int  lane    = tid & 31;
    const int  g       = lane >> 4;
    const int  ln      = lane & 15;
    const long rowBase = (long)blockIdx.x * ROWS_PER_BLOCK + (long)wave * ROWS_PER_WAVE;

    h16*   myX  = sX + wave * ROWS_PER_WAVE * KPAD;
    h16*   myH  = sH + wave * ROWS_PER_WAVE * KPAD;
    float* myLg = sLg + wave * ROWS_PER_WAVE * 32;
    float* myGs = sGs + wave * ROWS_PER_WAVE * N_ACT;

    // ---- stage x tile: f32 -> f16, K padded 100 -> 128 with zeros ----
    for (int i = lane; i < ROWS_PER_WAVE * KPAD; i += 32) {
        const int  r   = i >> 7, k = i & 127;
        const long row = rowBase + r;
        float      v   = 0.0f;
        if (row < B && k < N_OBS) v = x[row * N_OBS + k];
        myX[i] = (h16)v;
    }
    // group-of-5 sums in f32 from original data (exact zero detection for the mask);
    // consecutive lanes read consecutive 5-float spans -> contiguous wave access, L2 hits.
    for (int i = lane; i < ROWS_PER_WAVE * N_ACT; i += 32) {
        const int r = i / N_ACT, grp = i % N_ACT;
        long row = rowBase + r;
        if (row >= B) row = B - 1;
        const float* p = x + row * N_OBS + grp * 5;
        myGs[i] = p[0] + p[1] + p[2] + p[3] + p[4];
    }
    __syncthreads();

    // ---- layer 1: x(32x128 f16) @ W1 -> h1 (leaky_relu) ----
    gemm_dual_hidden(myX, sW1T, sB1, myH, lane);
    __syncthreads();
    // ---- layer 2: h1 @ W2 -> h2 (leaky_relu), ping-pong back into myX ----
    gemm_dual_hidden(myH, sW2T, sB2, myX, lane);
    __syncthreads();

    // ---- layer 3: h2 @ W3 -> logits (f32, N padded 20 -> 32), into sH-aliased buffer ----
    {
        v16h a0[4], a1[4];
#pragma unroll
        for (int k = 0; k < 4; ++k) {
            a0[k] = load_a_frag(myX, ln, g, 32 * k);
            a1[k] = load_a_frag(myX + 16 * KPAD, ln, g, 32 * k);
        }
        for (int j = 0; j < 2; ++j) {
            v8f  c0 = {}, c1 = {};
            v16h bc = load_b_frag(sW3T, j * 16 + ln, g, 0);
#pragma unroll
            for (int k = 0; k < 4; ++k) {
                v16h bn = bc;
                if (k < 3) bn = load_b_frag(sW3T, j * 16 + ln, g, 32 * (k + 1));
                c0 = __builtin_amdgcn_wmma_f32_16x16x32_f16(false, a0[k], false, bc, (short)0,
                                                            c0, false, false);
                c1 = __builtin_amdgcn_wmma_f32_16x16x32_f16(false, a1[k], false, bc, (short)0,
                                                            c1, false, false);
                bc = bn;
            }
            const int   N  = j * 16 + ln;
            const float bv = sB3[N];
#pragma unroll
            for (int r = 0; r < 8; ++r) {
                const int M = r + 8 * g;
                myLg[M * 32 + N]        = c0[r] + bv;
                myLg[(16 + M) * 32 + N] = c1[r] + bv;
            }
        }
    }
    __syncthreads();

    // ---- mask + softmax: each lane owns one of the wave's 32 rows ----
    {
        const long row = rowBase + lane;
        if (row < B) {
            float vals[N_ACT];
            float mx = -INFINITY;
#pragma unroll
            for (int a = 0; a < N_ACT; ++a) {
                float lg = myLg[lane * 32 + a];
                if (myGs[lane * N_ACT + a] == 0.0f) lg = -INFINITY;
                vals[a] = lg;
                mx      = fmaxf(mx, lg);
            }
            float s = 0.0f;
#pragma unroll
            for (int a = 0; a < N_ACT; ++a) {
                const float e = __expf(vals[a] - mx);
                vals[a]       = e;
                s += e;
            }
            const float inv = 1.0f / s;
            float* o = out + row * N_ACT;
#pragma unroll
            for (int a = 0; a < N_ACT; ++a) o[a] = vals[a] * inv;
        }
    }
}

extern "C" void kernel_launch(void* const* d_in, const int* in_sizes, int n_in,
                              void* d_out, int out_size, void* d_ws, size_t ws_size,
                              hipStream_t stream) {
    const float* x  = (const float*)d_in[0];
    const float* W1 = (const float*)d_in[1];
    const float* b1 = (const float*)d_in[2];
    const float* W2 = (const float*)d_in[3];
    const float* b2 = (const float*)d_in[4];
    const float* W3 = (const float*)d_in[5];
    const float* b3 = (const float*)d_in[6];
    float*       out = (float*)d_out;

    const int B      = in_sizes[0] / N_OBS;
    const int blocks = (B + ROWS_PER_BLOCK - 1) / ROWS_PER_BLOCK;

    dqn_fused<<<blocks, 256, LDS_TOTAL, stream>>>(x, W1, b1, W2, b2, W3, b3, out, B);
}
